// PGCoverageDecoder_9826885173542
// MI455X (gfx1250) — matrix operand order repaired and checked by reading, not compiled, not run on hardware
//
#include <hip/hip_runtime.h>
#include <hip/hip_bf16.h>
#include <math.h>

#define BB 16
#define SS 400
#define TT 51
#define NSTEP (TT - 1)        // 50
#define VV 50000
#define EE 128
#define HH 256
#define MAX_OOV 50
#define EXTV (VV + MAX_OOV)   // 50050
#define H4 (4 * HH)           // 1024
#define KDIM (2 * HH)         // 512  (concat(h, context))
#define XDIM (EE + HH)        // 384
#define PGDIM (2 * HH + EE)   // 640
#define NTILE (VV / 16)       // 3125 vocab tiles of 16
#define NPAIR ((NTILE + 1) / 2)

typedef __attribute__((ext_vector_type(16))) __bf16   v16bf;
typedef __attribute__((ext_vector_type(8)))  float    v8f;
typedef __attribute__((ext_vector_type(4)))  unsigned int u32x4;

union Frag { u32x4 u[2]; v16bf v; };

__device__ __forceinline__ unsigned short f2bf(float x) {
  unsigned u = __float_as_uint(x);
  unsigned r = u + 0x7FFFu + ((u >> 16) & 1u);      // round-to-nearest-even
  if ((u & 0x7F800000u) == 0x7F800000u) r = u;      // inf/nan pass-through
  return (unsigned short)(r >> 16);
}
__device__ __forceinline__ float sigmf(float x) { return 1.0f / (1.0f + expf(-x)); }

// ---------------- setup kernels ----------------

__global__ void k_init(const float* hidden, const float* cell,
                       const int* src_ids, const int* src_oov,
                       float* h, float* c, float* cov, int* cidx, float* accum) {
  int idx = blockIdx.x * 256 + threadIdx.x;
  if (idx < BB * SS) {
    cov[idx] = 0.0f;
    int o = src_oov[idx];
    cidx[idx] = (o >= 0) ? (VV + o) : src_ids[idx];
  }
  if (idx < BB * HH) { h[idx] = hidden[idx]; c[idx] = cell[idx]; }
  if (idx == 0) accum[0] = 0.0f;
}

// fp32 W_out is read exactly once per call: non-temporal load keeps it out of L2
// (the bf16 copy we produce is what must stay L2-resident for 50 GEMM steps).
__global__ void k_wout_bf16(const float* W, unsigned short* Wb) {
  size_t idx = (size_t)blockIdx.x * 256 + threadIdx.x;
  if (idx < (size_t)VV * KDIM) {
    float x = __builtin_nontemporal_load(&W[idx]);
    Wb[idx] = f2bf(x);
  }
}

// Wh_h[b,s,k] = sum_h enc[b,s,h] * W_h[k,h]   (one block per (b,s), thread = k)
__global__ void k_whh(const float* enc, const float* W_h, float* Whh) {
  __shared__ float row[HH];
  int bs = blockIdx.x;
  int t  = threadIdx.x;
  row[t] = enc[(size_t)bs * HH + t];
  __syncthreads();
  const float* w = W_h + (size_t)t * HH;
  float acc = 0.0f;
  for (int i = 0; i < HH; ++i) acc += w[i] * row[i];
  Whh[(size_t)bs * HH + t] = acc;
}

// ---------------- per-step kernels ----------------

__global__ void k_emb(const int* tgt, const float* embedding, float* emb, int step) {
  int idx = blockIdx.x * 256 + threadIdx.x;
  if (idx >= BB * EE) return;
  int b = idx / EE, e = idx % EE;
  int y = tgt[b * TT + step];
  emb[idx] = embedding[(size_t)y * EE + e];
}

// sW[b,k] = sum_h h[b,h] * W_s[k,h]
__global__ void k_sW(const float* h, const float* W_s, float* sW) {
  int idx = blockIdx.x * 256 + threadIdx.x;
  if (idx >= BB * HH) return;
  int b = idx >> 8, k = idx & (HH - 1);
  const float* w  = W_s + (size_t)k * HH;
  const float* hr = h + b * HH;
  float acc = 0.0f;
  for (int i = 0; i < HH; ++i) acc += hr[i] * w[i];
  sW[idx] = acc;
}

// one wave per (b,s): scores[b,s] = v . tanh(Wh_h + sW + cov*w_c + b_attn), masked
__global__ void k_score(const float* Whh, const float* sW, const float* cov,
                        const float* w_c, const float* vvec, const float* b_attn,
                        const int* src_lens, float* scores) {
  int gid  = blockIdx.x * 8 + (threadIdx.x >> 5);
  int lane = threadIdx.x & 31;
  if (gid >= BB * SS) return;
  int b = gid / SS, s = gid % SS;
  float out;
  if (s >= src_lens[b]) {
    out = -1e9f;
  } else {
    float cv = cov[b * SS + s];
    float ba = b_attn[0];
    const float* wrow = Whh + (size_t)(b * SS + s) * HH;
    float acc = 0.0f;
    for (int hh = lane; hh < HH; hh += 32) {
      float t = wrow[hh] + sW[b * HH + hh] + cv * w_c[hh] + ba;
      acc += vvec[hh] * tanhf(t);
    }
    for (int m = 16; m >= 1; m >>= 1) acc += __shfl_xor(acc, m, 32);
    out = acc;
  }
  if (lane == 0) scores[gid] = out;
}

// per-b: softmax over S, coverage loss, cov update, context = attn . enc
__global__ void k_attn_ctx(const float* scores, const float* enc, float* attn,
                           float* cov, float* ctx, float* accum) {
  __shared__ float sc[SS];
  __shared__ float red[256];
  int b = blockIdx.x, tid = threadIdx.x;
  for (int s = tid; s < SS; s += 256) sc[s] = scores[b * SS + s];
  __syncthreads();
  float mx = -3.4e38f;
  for (int s = tid; s < SS; s += 256) mx = fmaxf(mx, sc[s]);
  red[tid] = mx; __syncthreads();
  for (int s = 128; s >= 1; s >>= 1) { if (tid < s) red[tid] = fmaxf(red[tid], red[tid + s]); __syncthreads(); }
  mx = red[0]; __syncthreads();
  float sm = 0.0f;
  for (int s = tid; s < SS; s += 256) { float e = expf(sc[s] - mx); sc[s] = e; sm += e; }
  red[tid] = sm; __syncthreads();
  for (int s = 128; s >= 1; s >>= 1) { if (tid < s) red[tid] += red[tid + s]; __syncthreads(); }
  float denom = red[0]; __syncthreads();
  float lmin = 0.0f;
  for (int s = tid; s < SS; s += 256) {
    float a = sc[s] / denom;
    sc[s] = a;
    attn[b * SS + s] = a;
    float co = cov[b * SS + s];
    lmin += fminf(a, co);
    cov[b * SS + s] = co + a;
  }
  red[tid] = lmin; __syncthreads();
  for (int s = 128; s >= 1; s >>= 1) { if (tid < s) red[tid] += red[tid + s]; __syncthreads(); }
  if (tid == 0) atomicAdd(accum, red[0]);
  __syncthreads();
  // context[b, tid]
  float acc = 0.0f;
  for (int s = 0; s < SS; ++s) acc += sc[s] * enc[((size_t)b * SS + s) * HH + tid];
  ctx[b * HH + tid] = acc;
}

// gates[b,j] = x.W_ih[j] + h.W_hh[j] + b_ih + b_hh,  x = [emb | context]
__global__ void k_gates(const float* emb, const float* ctx, const float* h,
                        const float* W_ih, const float* W_hh,
                        const float* b_ih, const float* b_hh, float* gates) {
  int idx = blockIdx.x * 256 + threadIdx.x;
  if (idx >= BB * H4) return;
  int b = idx >> 10, j = idx & (H4 - 1);
  float acc = b_ih[j] + b_hh[j];
  const float* wi = W_ih + (size_t)j * XDIM;
  for (int k = 0; k < EE; ++k) acc += emb[b * EE + k] * wi[k];
  for (int k = 0; k < HH; ++k) acc += ctx[b * HH + k] * wi[EE + k];
  const float* wh = W_hh + (size_t)j * HH;
  for (int k = 0; k < HH; ++k) acc += h[b * HH + k] * wh[k];
  gates[idx] = acc;
}

// LSTM state update + pack A = bf16([h_new | context]) for the WMMA GEMM
__global__ void k_update(const float* gates, const float* ctx,
                         float* h, float* c, unsigned short* Abf) {
  int idx = blockIdx.x * 256 + threadIdx.x;
  if (idx >= BB * HH) return;
  int b = idx >> 8, q = idx & (HH - 1);
  const float* g = gates + (size_t)b * H4;
  float ig = sigmf(g[q]);
  float fg = sigmf(g[HH + q]);
  float gg = tanhf(g[2 * HH + q]);
  float og = sigmf(g[3 * HH + q]);
  float cn = fg * c[idx] + ig * gg;
  float hn = og * tanhf(cn);
  c[idx] = cn;
  h[idx] = hn;
  Abf[(size_t)b * KDIM + q]      = f2bf(hn);
  Abf[(size_t)b * KDIM + HH + q] = f2bf(ctx[idx]);
}

// one wave per b: p_gen = sigmoid(W_pg . [h_new | context | emb] + b_pg)
__global__ void k_pgen(const float* h, const float* ctx, const float* emb,
                       const float* W_pg, const float* b_pg, float* pgen) {
  int b = threadIdx.x >> 5, lane = threadIdx.x & 31;
  float acc = 0.0f;
  for (int k = lane; k < PGDIM; k += 32) {
    float z = (k < HH) ? h[b * HH + k]
            : (k < 2 * HH) ? ctx[b * HH + k - HH]
            : emb[b * EE + k - 2 * HH];
    acc += W_pg[k] * z;
  }
  for (int m = 16; m >= 1; m >>= 1) acc += __shfl_xor(acc, m, 32);
  if (lane == 0) pgen[b] = sigmf(acc + b_pg[0]);
}

// ---- WMMA vocab projection ----
// logits[16, V] = A(bf16)[16,512] x W_out(bf16)^T + b_out via v_wmma_f32_16x16x32_bf16.
// Each wave owns TWO 16x16 (batch x vocab) tiles sharing one A fragment per K-step:
// 6 b128 loads per 2 WMMA ops instead of 8, and A traffic halved.

__device__ __forceinline__ void store_tile(float* __restrict__ logits,
                                           const float* __restrict__ b_out,
                                           v8f acc, int tile, int r, int hi) {
  int v0 = tile * 16;
  int mb = hi * 8;
  float bo = b_out[v0 + r];
  #pragma unroll
  for (int q = 0; q < 8; ++q)
    logits[(size_t)(mb + q) * VV + v0 + r] = acc[q] + bo;
}

__global__ void k_logits(const unsigned short* __restrict__ Abf,
                         const unsigned short* __restrict__ Wb,
                         const float* __restrict__ b_out,
                         float* __restrict__ logits) {
  int pair = blockIdx.x * 8 + (threadIdx.x >> 5);
  int t0 = pair * 2;
  if (t0 >= NTILE) return;                   // wave-uniform: EXEC all-1s inside
  int lane = threadIdx.x & 31;
  int r  = lane & 15;                        // A: row m ; B: column n ; D: column n
  int hi = lane >> 4;                        // half select
  const unsigned short* aBase = Abf + (size_t)r * KDIM + hi * 8;        // A: K 0..7/8..15 (+16)
  const unsigned short* bBase = Wb + (size_t)(t0 * 16 + r) * KDIM + hi * 8;
  v8f acc0 = {0.f, 0.f, 0.f, 0.f, 0.f, 0.f, 0.f, 0.f};
  if (t0 + 1 < NTILE) {                      // wave-uniform branch
    const unsigned short* bBase1 = bBase + (size_t)16 * KDIM;
    v8f acc1 = {0.f, 0.f, 0.f, 0.f, 0.f, 0.f, 0.f, 0.f};
    for (int k0 = 0; k0 < KDIM; k0 += 32) {
      Frag fa, fb0, fb1;
      fa.u[0]  = *(const u32x4*)(aBase  + k0);
      fa.u[1]  = *(const u32x4*)(aBase  + k0 + 16);
      fb0.u[0] = *(const u32x4*)(bBase  + k0);
      fb0.u[1] = *(const u32x4*)(bBase  + k0 + 16);
      fb1.u[0] = *(const u32x4*)(bBase1 + k0);
      fb1.u[1] = *(const u32x4*)(bBase1 + k0 + 16);
      acc0 = __builtin_amdgcn_wmma_f32_16x16x32_bf16(false, fa.v, false, fb0.v,
                                                     (short)0, acc0, false, false);
      acc1 = __builtin_amdgcn_wmma_f32_16x16x32_bf16(false, fa.v, false, fb1.v,
                                                     (short)0, acc1, false, false);
    }
    store_tile(logits, b_out, acc0, t0, r, hi);
    store_tile(logits, b_out, acc1, t0 + 1, r, hi);
  } else {                                   // odd tail tile
    for (int k0 = 0; k0 < KDIM; k0 += 32) {
      Frag fa, fb0;
      fa.u[0]  = *(const u32x4*)(aBase + k0);
      fa.u[1]  = *(const u32x4*)(aBase + k0 + 16);
      fb0.u[0] = *(const u32x4*)(bBase + k0);
      fb0.u[1] = *(const u32x4*)(bBase + k0 + 16);
      acc0 = __builtin_amdgcn_wmma_f32_16x16x32_bf16(false, fa.v, false, fb0.v,
                                                     (short)0, acc0, false, false);
    }
    store_tile(logits, b_out, acc0, t0, r, hi);
  }
}

// per-b: max and sum(exp) over V
__global__ void k_vreduce(const float* logits, float* rmax, float* rsum) {
  __shared__ float red[512];
  int b = blockIdx.x, tid = threadIdx.x;
  const float* row = logits + (size_t)b * VV;
  float mx = -3.4e38f;
  for (int i = tid; i < VV; i += 512) mx = fmaxf(mx, row[i]);
  red[tid] = mx; __syncthreads();
  for (int s = 256; s >= 1; s >>= 1) { if (tid < s) red[tid] = fmaxf(red[tid], red[tid + s]); __syncthreads(); }
  mx = red[0]; __syncthreads();
  float sm = 0.0f;
  for (int i = tid; i < VV; i += 512) sm += expf(row[i] - mx);
  red[tid] = sm; __syncthreads();
  for (int s = 256; s >= 1; s >>= 1) { if (tid < s) red[tid] += red[tid + s]; __syncthreads(); }
  if (tid == 0) { rmax[b] = mx; rsum[b] = red[0]; }
}

// final[b, step, v] = p_gen * softmax(logits) padded to EXT (copy dist added after).
// Output is write-once and never re-read on device: non-temporal stores keep the
// 160MB of distributions from evicting the L2-resident bf16 W_out working set.
__global__ void k_fdist(const float* logits, const float* rmax, const float* rsum,
                        const float* pgen, float* out, int step) {
  int idx = blockIdx.x * 256 + threadIdx.x;
  if (idx >= BB * EXTV) return;
  int b = idx / EXTV, vv = idx % EXTV;
  float val = 0.0f;
  if (vv < VV)
    val = pgen[b] * expf(logits[(size_t)b * VV + vv] - rmax[b]) / rsum[b];
  __builtin_nontemporal_store(val, &out[((size_t)b * NSTEP + step) * EXTV + vv]);
}

// scatter copy distribution: out[b, step, copy_idx] += (1-p_gen)*attn
__global__ void k_scatter(const float* attn, const float* pgen, const int* cidx,
                          float* out, int step) {
  int idx = blockIdx.x * 256 + threadIdx.x;
  if (idx >= BB * SS) return;
  int b = idx / SS;
  float w = (1.0f - pgen[b]) * attn[idx];
  atomicAdd(&out[((size_t)b * NSTEP + step) * EXTV + cidx[idx]], w);
}

__global__ void k_finish(const float* h, const float* c, const float* accum, float* out) {
  int idx = blockIdx.x * 256 + threadIdx.x;
  size_t base = (size_t)BB * NSTEP * EXTV;
  if (idx < BB * HH) {
    out[base + idx]           = h[idx];
    out[base + BB * HH + idx] = c[idx];
  }
  if (idx == 0)
    out[base + 2 * BB * HH] = accum[0] * (1.0f / (float)(NSTEP * BB)); // COV_LOSS_WEIGHT=1
}

// ---------------- host side ----------------

extern "C" void kernel_launch(void* const* d_in, const int* in_sizes, int n_in,
                              void* d_out, int out_size, void* d_ws, size_t ws_size,
                              hipStream_t stream) {
  (void)in_sizes; (void)n_in; (void)out_size; (void)ws_size;
  const int*   tgt       = (const int*)d_in[0];
  const float* hidden    = (const float*)d_in[1];
  const float* cell      = (const float*)d_in[2];
  const float* enc       = (const float*)d_in[3];
  const int*   src_lens  = (const int*)d_in[4];
  const int*   src_ids   = (const int*)d_in[5];
  const int*   src_oov   = (const int*)d_in[6];
  const float* embedding = (const float*)d_in[7];
  const float* W_h       = (const float*)d_in[8];
  const float* W_s       = (const float*)d_in[9];
  const float* w_c       = (const float*)d_in[10];
  const float* vvec      = (const float*)d_in[11];
  const float* b_attn    = (const float*)d_in[12];
  const float* W_ih      = (const float*)d_in[13];
  const float* W_hh      = (const float*)d_in[14];
  const float* b_ih      = (const float*)d_in[15];
  const float* b_hh      = (const float*)d_in[16];
  const float* W_pg      = (const float*)d_in[17];
  const float* b_pg      = (const float*)d_in[18];
  const float* W_out     = (const float*)d_in[19];
  const float* b_out     = (const float*)d_in[20];
  float* out = (float*)d_out;

  size_t off = 0;
  auto alloc = [&](size_t bytes) -> void* {
    void* p = (char*)d_ws + off;
    off += (bytes + 255) & ~(size_t)255;
    return p;
  };
  float*          Whh    = (float*)alloc((size_t)BB * SS * HH * 4);
  unsigned short* Wb     = (unsigned short*)alloc((size_t)VV * KDIM * 2);
  float*          logits = (float*)alloc((size_t)BB * VV * 4);
  unsigned short* Abf    = (unsigned short*)alloc((size_t)BB * KDIM * 2);
  float*          hS     = (float*)alloc(BB * HH * 4);
  float*          cS     = (float*)alloc(BB * HH * 4);
  float*          sW     = (float*)alloc(BB * HH * 4);
  float*          cov    = (float*)alloc(BB * SS * 4);
  float*          attn   = (float*)alloc(BB * SS * 4);
  float*          scores = (float*)alloc(BB * SS * 4);
  float*          ctx    = (float*)alloc(BB * HH * 4);
  float*          emb    = (float*)alloc(BB * EE * 4);
  float*          gates  = (float*)alloc(BB * H4 * 4);
  float*          pgen   = (float*)alloc(256);
  float*          rmax   = (float*)alloc(256);
  float*          rsum   = (float*)alloc(256);
  int*            cidx   = (int*)alloc(BB * SS * 4);
  float*          accum  = (float*)alloc(256);

  // loop-invariant setup
  k_init<<<(BB * SS + 255) / 256, 256, 0, stream>>>(hidden, cell, src_ids, src_oov,
                                                    hS, cS, cov, cidx, accum);
  k_wout_bf16<<<(VV * KDIM + 255) / 256, 256, 0, stream>>>(W_out, Wb);
  k_whh<<<BB * SS, HH, 0, stream>>>(enc, W_h, Whh);

  for (int step = 0; step < NSTEP; ++step) {
    k_emb<<<(BB * EE + 255) / 256, 256, 0, stream>>>(tgt, embedding, emb, step);
    k_sW<<<(BB * HH + 255) / 256, 256, 0, stream>>>(hS, W_s, sW);
    k_score<<<(BB * SS + 7) / 8, 256, 0, stream>>>(Whh, sW, cov, w_c, vvec, b_attn,
                                                   src_lens, scores);
    k_attn_ctx<<<BB, 256, 0, stream>>>(scores, enc, attn, cov, ctx, accum);
    k_gates<<<(BB * H4 + 255) / 256, 256, 0, stream>>>(emb, ctx, hS, W_ih, W_hh,
                                                       b_ih, b_hh, gates);
    k_update<<<(BB * HH + 255) / 256, 256, 0, stream>>>(gates, ctx, hS, cS, Abf);
    k_pgen<<<1, 512, 0, stream>>>(hS, ctx, emb, W_pg, b_pg, pgen);
    k_logits<<<(NPAIR + 7) / 8, 256, 0, stream>>>(Abf, Wb, b_out, logits);
    k_vreduce<<<BB, 512, 0, stream>>>(logits, rmax, rsum);
    k_fdist<<<(BB * EXTV + 255) / 256, 256, 0, stream>>>(logits, rmax, rsum, pgen,
                                                         out, step);
    k_scatter<<<(BB * SS + 255) / 256, 256, 0, stream>>>(attn, pgen, cidx, out, step);
  }
  k_finish<<<(BB * HH + 255) / 256, 256, 0, stream>>>(hS, cS, accum, out);
}
